// GridLSTM_53300544143495
// MI455X (gfx1250) — matrix operand order repaired and checked by reading, not compiled, not run on hardware
//
#include <hip/hip_runtime.h>
#include <hip/hip_bf16.h>
#include <math.h>

typedef __attribute__((ext_vector_type(16))) _Float16 v16h;
typedef __attribute__((ext_vector_type(8)))  float    v8f;

union Frag16 { v16h v; unsigned int u[8]; };

// ---------------------------------------------------------------------------
// WMMA helpers (CDNA5 wave32, v_wmma_f32_16x16x32_f16)
// ---------------------------------------------------------------------------
__device__ inline v8f wmma_f16(v16h a, v16h b, v8f c) {
    return __builtin_amdgcn_wmma_f32_16x16x32_f16(
        /*neg_a*/false, a, /*neg_b*/false, b,
        /*c_mod*/(short)0, c, /*reuse_a*/false, /*reuse_b*/false);
}

// A-fragment: 16x32 f16 tile from row-major [., ld] matrix starting (row0,k0).
// contiguous k order (same as B): m = lane&15; lanes 0-15 hold K 0..15, lanes 16-31 hold
// K 16..31; each VGPR = 2 packed f16 (vectorizes to 2x b128 loads).
__device__ inline v16h load_a_frag(const _Float16* A, int ld, int row0, int k0, int lid) {
    Frag16 f;
    const _Float16* rowp = A + (row0 + (lid & 15)) * ld + k0;
    int khalf = (lid >> 4) * 16;
#pragma unroll
    for (int v = 0; v < 8; ++v) {
        int k = khalf + 2 * v;
        f.u[v] = *(const unsigned int*)(rowp + k);
    }
    return f.v;
}

// B-fragment: 32x16 f16 tile where B[k][n] = W[n][k], W row-major [N,ldk].
// n = lane&15; lanes 0-15 hold K 0..15, lanes 16-31 hold K 16..31.
__device__ inline v16h load_bT_frag(const _Float16* W, int ldk, int n0, int k0, int lid) {
    Frag16 f;
    const _Float16* rowp = W + (n0 + (lid & 15)) * ldk + k0 + (lid >> 4) * 16;
#pragma unroll
    for (int v = 0; v < 8; ++v)
        f.u[v] = *(const unsigned int*)(rowp + 2 * v);
    return f.v;
}

__device__ inline float sigmoidf_(float x) { return 1.0f / (1.0f + expf(-x)); }

// ---------------------------------------------------------------------------
// Problem constants
// ---------------------------------------------------------------------------
#define LL 32
#define BB 8
#define HH 128
#define EE 300
#define EPAD 320
#define NG 1024  // 8*H gate width
#define HH2 256  // 2*H

// ---------------------------------------------------------------------------
// Workspace layout (bytes, all naturally 256B aligned)
// ---------------------------------------------------------------------------
constexpr size_t SZ_XP   = (size_t)256 * EPAD * 2;        // f16 [256,320]
constexpr size_t SZ_W16  = (size_t)NG * EPAD * 2;         // f16 [1024,320]
constexpr size_t SZ_WHH  = (size_t)NG * HH2 * 2;          // f16 [1024,256]
constexpr size_t SZ_WC   = (size_t)HH * HH2 * 2;          // f16 [128,256]
constexpr size_t SZ_G    = (size_t)256 * NG * 4;          // f32 [256,1024]
constexpr size_t SZ_HG   = (size_t)LL * LL * BB * HH * 2; // f16 grid h
constexpr size_t SZ_CG   = (size_t)LL * LL * BB * HH * 4; // f32 grid c
constexpr size_t SZ_FLG  = 256;                           // progress flags

constexpr size_t OFF_XP   = 0;
constexpr size_t OFF_XH   = OFF_XP + SZ_XP;
constexpr size_t OFF_WP   = OFF_XH + SZ_XP;
constexpr size_t OFF_WH   = OFF_WP + SZ_W16;
constexpr size_t OFF_WHH  = OFF_WH + SZ_W16;
constexpr size_t OFF_WCH  = OFF_WHH + SZ_WHH;
constexpr size_t OFF_WCC  = OFF_WCH + SZ_WC;
constexpr size_t OFF_GP   = OFF_WCC + SZ_WC;
constexpr size_t OFF_GH   = OFF_GP + SZ_G;
constexpr size_t OFF_HG   = OFF_GH + SZ_G;
constexpr size_t OFF_CG   = OFF_HG + SZ_HG;
constexpr size_t OFF_FLG  = OFF_CG + SZ_CG;

// ---------------------------------------------------------------------------
// Prep: gather embeddings into padded f16 activation matrices
// ---------------------------------------------------------------------------
__global__ void k_gather(const int* premise, const int* hypothesis, const float* emb,
                         _Float16* Xp, _Float16* Xh, int* flags) {
    int idx = blockIdx.x * blockDim.x + threadIdx.x;
    if (idx < LL) flags[idx] = 0;  // zero per-column progress flags each call
    if (idx >= 2 * 256 * EPAD) return;
    int which = idx / (256 * EPAD);
    int rem   = idx % (256 * EPAD);
    int row = rem / EPAD, k = rem % EPAD;
    int seq = row >> 3, b = row & 7;
    const int* tokp = which ? hypothesis : premise;
    int tok = tokp[b * LL + seq];
    float v = (k < EE) ? emb[(size_t)tok * EE + k] : 0.0f;
    (which ? Xh : Xp)[rem] = (_Float16)v;
}

// ---------------------------------------------------------------------------
// Prep: convert / split weights to f16
// ---------------------------------------------------------------------------
__global__ void k_wprep(const float* W_ih, const float* W_hh, const float* W_ch,
                        const float* W_cc, _Float16* Wp16, _Float16* Wh16,
                        _Float16* Whh16, _Float16* Wch16, _Float16* Wcc16) {
    const int S0 = NG * EPAD, S1 = 2 * S0, S2 = S1 + NG * HH2,
              S3 = S2 + HH * HH2, S4 = S3 + HH * HH2;
    int idx = blockIdx.x * blockDim.x + threadIdx.x;
    if (idx < S0) {
        int r = idx / EPAD, k = idx % EPAD;
        Wp16[idx] = (_Float16)(k < EE ? W_ih[r * 600 + k] : 0.0f);
    } else if (idx < S1) {
        int t = idx - S0; int r = t / EPAD, k = t % EPAD;
        Wh16[t] = (_Float16)(k < EE ? W_ih[r * 600 + EE + k] : 0.0f);
    } else if (idx < S2) {
        int t = idx - S1; Whh16[t] = (_Float16)W_hh[t];
    } else if (idx < S3) {
        int t = idx - S2; Wch16[t] = (_Float16)W_ch[t];
    } else if (idx < S4) {
        int t = idx - S3; Wcc16[t] = (_Float16)W_cc[t];
    }
}

// ---------------------------------------------------------------------------
// Phase 1: G_p = X_p @ Wp^T + (b_ih+b_hh), G_h = X_h @ Wh^T  (M=256,N=1024,K=320)
// ---------------------------------------------------------------------------
__global__ __launch_bounds__(256) void k_phase1(const _Float16* Xp, const _Float16* Xh,
                                                const _Float16* Wp16, const _Float16* Wh16,
                                                const float* b_ih, const float* b_hh,
                                                float* Gp, float* Gh) {
    int gw = (blockIdx.x * 256 + threadIdx.x) >> 5;
    int lid = threadIdx.x & 31;
    int matrix = gw >> 10;
    int tileId = gw & 1023;
    int mt = tileId >> 6, nt = tileId & 63;
    const _Float16* X = matrix ? Xh : Xp;
    const _Float16* W = matrix ? Wh16 : Wp16;
    float* G = matrix ? Gh : Gp;

    v8f acc = {};
#pragma unroll
    for (int kt = 0; kt < 10; ++kt) {
        v16h a = load_a_frag(X, EPAD, mt * 16, kt * 32, lid);
        v16h b = load_bT_frag(W, EPAD, nt * 16, kt * 32, lid);
        acc = wmma_f16(a, b, acc);
    }
    int n = nt * 16 + (lid & 15);
    int mrow = mt * 16 + (lid >> 4) * 8;
    float bias = matrix ? 0.0f : (b_ih[n] + b_hh[n]);
#pragma unroll
    for (int r = 0; r < 8; ++r)
        G[(size_t)(mrow + r) * NG + n] = acc[r] + bias;
}

// ---------------------------------------------------------------------------
// Persistent wavefront kernel: one workgroup per column j, walks rows i=0..31.
//   up-deps (i-1,j) live in LDS; left-deps (i,j-1) via global + acquire flags.
//   Wave w owns channel-tiles {w, w+8} for all 4 gates -> LSTM nonlinearity
//   runs in registers (i,f,g,o share lane/VGPR slots), no gates LDS buffer.
// ---------------------------------------------------------------------------
__global__ __launch_bounds__(256) void k_grid(const float* Gp, const float* Gh,
                                              const _Float16* Whh16, const _Float16* Wch16,
                                              const _Float16* Wcc16, const float* b_ch,
                                              const float* b_cc, _Float16* Hgrid,
                                              float* Cgrid, int* flags) {
    __shared__ _Float16 hin[16][HH2];  // [h_up , h_left], rows 8..15 zero
    __shared__ _Float16 ho[16][HH2];   // h_o (K for condense), rows 8..15 zero
    __shared__ _Float16 co[16][HH2];   // c_o
    __shared__ _Float16 hup[BB][HH];   // condensed h of row i-1 (own column)
    __shared__ float    cup[BB][HH];   // condensed c of row i-1 (own column)

    const int j = blockIdx.x;
    const int tid = threadIdx.x, wave = tid >> 5, lid = tid & 31;

    for (int e = tid; e < 16 * HH2; e += 256) {
        int m = e >> 8, k = e & 255;
        hin[m][k] = (_Float16)0.0f;
        ho[m][k]  = (_Float16)0.0f;
        co[m][k]  = (_Float16)0.0f;
    }
    for (int e = tid; e < BB * HH; e += 256) {
        hup[e >> 7][e & 127] = (_Float16)0.0f;
        cup[e >> 7][e & 127] = 0.0f;
    }
    __syncthreads();

    for (int i = 0; i < LL; ++i) {
        // ---- wait for left column to publish row i ----
        if (j > 0) {
            if (tid == 0) {
                while (__hip_atomic_load(&flags[j - 1], __ATOMIC_ACQUIRE,
                                         __HIP_MEMORY_SCOPE_AGENT) <= i)
                    __builtin_amdgcn_s_sleep(1);
            }
            __syncthreads();
        }
        const _Float16* hl = (j > 0) ? (Hgrid + (size_t)((i * LL + (j - 1)) * BB) * HH) : nullptr;
        const float*    cl = (j > 0) ? (Cgrid + (size_t)((i * LL + (j - 1)) * BB) * HH) : nullptr;

        // ---- stage h_in rows 0..7 (up half from LDS, left half from global) ----
        for (int e = tid; e < BB * HH2; e += 256) {
            int m = e >> 8, k = e & 255;
            hin[m][k] = (k < HH) ? hup[m][k]
                                 : (hl ? hl[m * HH + (k - HH)] : (_Float16)0.0f);
        }
        __syncthreads();

        // ---- gates GEMM: acc[t][g] = tile (channel-tile wave+8t, gate g) ----
        v8f acc[2][4] = {};
#pragma unroll
        for (int kt = 0; kt < 8; ++kt) {
            v16h a = load_a_frag(&hin[0][0], HH2, 0, kt * 32, lid);  // shared across tiles
#pragma unroll
            for (int t = 0; t < 2; ++t) {
#pragma unroll
                for (int g = 0; g < 4; ++g) {
                    int n0 = g * HH2 + (wave + 8 * t) * 16;
                    v16h b = load_bT_frag(Whh16, HH2, n0, kt * 32, lid);
                    acc[t][g] = wmma_f16(a, b, acc[t][g]);
                }
            }
        }

        // ---- LSTM nonlinearity in registers (lanes 0..15 = batch rows) ----
        const float* GpR = Gp + (size_t)(i * BB) * NG;
        const float* GhR = Gh + (size_t)(j * BB) * NG;
        if (lid < 16) {
#pragma unroll
            for (int t = 0; t < 2; ++t) {
                int nc = (wave + 8 * t) * 16 + lid;  // channel 0..255
#pragma unroll
                for (int r = 0; r < 8; ++r) {
                    float gi = acc[t][0][r] + GpR[r * NG + 0 * HH2 + nc] + GhR[r * NG + 0 * HH2 + nc];
                    float gf = acc[t][1][r] + GpR[r * NG + 1 * HH2 + nc] + GhR[r * NG + 1 * HH2 + nc];
                    float gg = acc[t][2][r] + GpR[r * NG + 2 * HH2 + nc] + GhR[r * NG + 2 * HH2 + nc];
                    float go = acc[t][3][r] + GpR[r * NG + 3 * HH2 + nc] + GhR[r * NG + 3 * HH2 + nc];
                    float cin = (t == 0) ? cup[r][nc]
                                         : (cl ? cl[r * HH + (nc - HH)] : 0.0f);
                    float cn = sigmoidf_(gf) * cin + sigmoidf_(gi) * tanhf(gg);
                    float hn = sigmoidf_(go) * tanhf(cn);
                    ho[r][nc] = (_Float16)hn;
                    co[r][nc] = (_Float16)cn;
                }
            }
        }
        __syncthreads();

        // ---- condense GEMMs: wave w owns output n-tile w for both h and c ----
        {
            int n0 = wave * 16;
            v8f acch = {}, accc = {};
#pragma unroll
            for (int kt = 0; kt < 8; ++kt) {
                v16h ah = load_a_frag(&ho[0][0], HH2, 0, kt * 32, lid);
                v16h bh = load_bT_frag(Wch16, HH2, n0, kt * 32, lid);
                acch = wmma_f16(ah, bh, acch);
                v16h ac = load_a_frag(&co[0][0], HH2, 0, kt * 32, lid);
                v16h bc = load_bT_frag(Wcc16, HH2, n0, kt * 32, lid);
                accc = wmma_f16(ac, bc, accc);
            }
            if (lid < 16) {
                int n = n0 + lid;
                _Float16* Ho = Hgrid + (size_t)((i * LL + j) * BB) * HH;
                float*    Co = Cgrid + (size_t)((i * LL + j) * BB) * HH;
#pragma unroll
                for (int r = 0; r < 8; ++r) {
                    float hv = acch[r] + b_ch[n];
                    float cv = accc[r] + b_cc[n];
                    Ho[r * HH + n] = (_Float16)hv;
                    Co[r * HH + n] = cv;
                    hup[r][n] = (_Float16)hv;  // feeds next row locally
                    cup[r][n] = cv;
                }
            }
        }

        // ---- publish row i to the right neighbor ----
        __threadfence();      // make Hgrid/Cgrid stores visible at agent scope
        __syncthreads();      // all threads' fences complete before flag store
        if (tid == 0)
            __hip_atomic_store(&flags[j], i + 1, __ATOMIC_RELEASE,
                               __HIP_MEMORY_SCOPE_AGENT);
    }
}

// ---------------------------------------------------------------------------
// Final MLP + softmax on corner state (tiny: scalar, single block)
// ---------------------------------------------------------------------------
__global__ __launch_bounds__(128) void k_mlp(const _Float16* Hgrid,
                                             const float* W1, const float* b1,
                                             const float* W2, const float* b2,
                                             const float* W3, const float* b3,
                                             float* out) {
    __shared__ float h0[BB][HH], h1[BB][HH];
    __shared__ float logits[BB][3];
    int tid = threadIdx.x;
    const _Float16* hc = Hgrid + (size_t)(((LL - 1) * LL + (LL - 1)) * BB) * HH;
    for (int e = tid; e < BB * HH; e += 128) h0[e >> 7][e & 127] = (float)hc[e];
    __syncthreads();
    for (int e = tid; e < BB * HH; e += 128) {
        int m = e >> 7, n = e & 127;
        float s = b1[n];
        for (int k = 0; k < HH; ++k) s += h0[m][k] * W1[n * HH + k];
        h1[m][n] = fmaxf(s, 0.0f);
    }
    __syncthreads();
    for (int e = tid; e < BB * HH; e += 128) {
        int m = e >> 7, n = e & 127;
        float s = b2[n];
        for (int k = 0; k < HH; ++k) s += h1[m][k] * W2[n * HH + k];
        h0[m][n] = fmaxf(s, 0.0f);
    }
    __syncthreads();
    if (tid < BB * 3) {
        int m = tid / 3, n = tid % 3;
        float s = b3[n];
        for (int k = 0; k < HH; ++k) s += h0[m][k] * W3[n * HH + k];
        logits[m][n] = s;
    }
    __syncthreads();
    if (tid < BB) {
        float a = logits[tid][0], b = logits[tid][1], c = logits[tid][2];
        float mx = fmaxf(a, fmaxf(b, c));
        float ea = expf(a - mx), eb = expf(b - mx), ec = expf(c - mx);
        float s = ea + eb + ec;
        out[tid * 3 + 0] = ea / s;
        out[tid * 3 + 1] = eb / s;
        out[tid * 3 + 2] = ec / s;
    }
}

// ---------------------------------------------------------------------------
// Host launcher (5 launches total; graph-capture safe)
// ---------------------------------------------------------------------------
extern "C" void kernel_launch(void* const* d_in, const int* in_sizes, int n_in,
                              void* d_out, int out_size, void* d_ws, size_t ws_size,
                              hipStream_t stream) {
    const int*   premise    = (const int*)d_in[0];
    const int*   hypothesis = (const int*)d_in[1];
    const float* emb        = (const float*)d_in[2];
    const float* W_ih       = (const float*)d_in[3];
    const float* b_ih       = (const float*)d_in[4];
    const float* W_hh       = (const float*)d_in[5];
    const float* b_hh       = (const float*)d_in[6];
    const float* W_ch       = (const float*)d_in[7];
    const float* b_ch       = (const float*)d_in[8];
    const float* W_cc       = (const float*)d_in[9];
    const float* b_cc       = (const float*)d_in[10];
    const float* W1         = (const float*)d_in[11];
    const float* b1         = (const float*)d_in[12];
    const float* W2         = (const float*)d_in[13];
    const float* b2         = (const float*)d_in[14];
    const float* W3         = (const float*)d_in[15];
    const float* b3         = (const float*)d_in[16];

    char* ws = (char*)d_ws;
    _Float16* Xp    = (_Float16*)(ws + OFF_XP);
    _Float16* Xh    = (_Float16*)(ws + OFF_XH);
    _Float16* Wp16  = (_Float16*)(ws + OFF_WP);
    _Float16* Wh16  = (_Float16*)(ws + OFF_WH);
    _Float16* Whh16 = (_Float16*)(ws + OFF_WHH);
    _Float16* Wch16 = (_Float16*)(ws + OFF_WCH);
    _Float16* Wcc16 = (_Float16*)(ws + OFF_WCC);
    float*    Gp    = (float*)(ws + OFF_GP);
    float*    Gh    = (float*)(ws + OFF_GH);
    _Float16* Hgrid = (_Float16*)(ws + OFF_HG);
    float*    Cgrid = (float*)(ws + OFF_CG);
    int*      flags = (int*)(ws + OFF_FLG);

    // 1) gather + f16 conversion of activations (also zeroes progress flags)
    {
        int total = 2 * 256 * EPAD;
        k_gather<<<(total + 255) / 256, 256, 0, stream>>>(premise, hypothesis, emb,
                                                          Xp, Xh, flags);
    }
    // 2) weight conversion / split / padding
    {
        int total = 2 * NG * EPAD + NG * HH2 + 2 * HH * HH2;
        k_wprep<<<(total + 255) / 256, 256, 0, stream>>>(W_ih, W_hh, W_ch, W_cc,
                                                         Wp16, Wh16, Whh16, Wch16, Wcc16);
    }
    // 3) hoisted input projections (WMMA GEMM)
    k_phase1<<<256, 256, 0, stream>>>(Xp, Xh, Wp16, Wh16, b_ih, b_hh, Gp, Gh);

    // 4) persistent pipelined wavefront: one workgroup per column
    k_grid<<<LL, 256, 0, stream>>>(Gp, Gh, Whh16, Wch16, Wcc16,
                                   b_ch, b_cc, Hgrid, Cgrid, flags);

    // 5) final MLP + softmax
    k_mlp<<<1, 128, 0, stream>>>(Hgrid, W1, b1, W2, b2, W3, b3, (float*)d_out);
}
